// SpatialFrequencyAttention_20452634263756
// MI455X (gfx1250) — matrix operand order repaired and checked by reading, not compiled, not run on hardware
//
#include <hip/hip_runtime.h>
#include <hip/hip_bf16.h>

// ---------------- problem constants ----------------
#define BATCH 16
#define CDIM  16
#define NWIN  25
#define HDIM  16
#define WDIM  128
#define CH    256           // c*h (t dimension)
#define BM    32            // rows of att per workgroup
#define NSLAB (CH / BM)     // 8 slabs per batch -> 128 workgroups
#define HW    (HDIM * WDIM) // 2048

typedef __bf16 bf16_t;
typedef bf16_t v4bf  __attribute__((ext_vector_type(4)));
typedef bf16_t v8bf  __attribute__((ext_vector_type(8)));
typedef bf16_t v16bf __attribute__((ext_vector_type(16)));
typedef float  v8f   __attribute__((ext_vector_type(8)));

// ---------------- LDS layout (byte offsets into dynamic smem) ----------------
// Phase 1:  Qs[32][136] bf16 @0 (8704B), Ks[256][136] bf16 @8704 (69632B) -> 78336
// Phase 2:  S[32][256] f32 @0 (32768B), P[32][264] bf16 @67584 (16896B)
// Phase 3:  Vt[128][264] bf16 @0 (67584B), P stays @67584
#define QS_STR 136
#define KS_STR 136
#define KS_OFF 8704
#define P_OFF  67584
#define P_STR  264
#define VT_STR 264
#define SMEM_BYTES 84480   // < 320KB/2 -> two workgroups can share a WGP

// -------- WMMA fragment loads from LDS (bf16, 16x16x32) --------
// A layout (ISA 7.12.2): m=lane&15, half=lane>>4; element e -> k = 16*(e>>3) + 8*half + (e&7)
__device__ __forceinline__ v16bf frag_a(const bf16_t* rowp, int half) {
  v8bf lo = *(const v8bf*)(rowp + 8 * half);
  v8bf hi = *(const v8bf*)(rowp + 16 + 8 * half);
  return __builtin_shufflevector(lo, hi, 0,1,2,3,4,5,6,7,8,9,10,11,12,13,14,15);
}
// B layout: n=lane&15; element e -> k = 16*(e>>3) + 8*half + (e&7)
__device__ __forceinline__ v16bf frag_b(const bf16_t* rowp, int half) {
  v8bf lo = *(const v8bf*)(rowp + 8 * half);
  v8bf hi = *(const v8bf*)(rowp + 16 + 8 * half);
  return __builtin_shufflevector(lo, hi, 0,1,2,3,4,5,6,7,8,9,10,11,12,13,14,15);
}

// ================= Kernel 1: inverse L2 norms of Q rows and K columns =================
__global__ void sfa_norms(const float* __restrict__ q, const float* __restrict__ k,
                          float* __restrict__ invq, float* __restrict__ invk) {
  const int b    = blockIdx.x >> 4;
  const int c    = blockIdx.x & 15;
  const int wid  = threadIdx.x >> 5;   // 0..7
  const int lane = threadIdx.x & 31;
  const size_t base = ((size_t)b * CDIM + c) * (NWIN * HW);

  for (int which = 0; which < 2; ++which) {
    const float* src = which ? k : q;
    float* dst       = which ? invk : invq;
    for (int hh = 0; hh < 2; ++hh) {
      const int h = wid * 2 + hh;
      float s = 0.f;
      for (int n = 0; n < NWIN; ++n) {
        float4 f = *(const float4*)(src + base + (size_t)n * HW + h * WDIM + lane * 4);
        s += f.x * f.x + f.y * f.y + f.z * f.z + f.w * f.w;
      }
#pragma unroll
      for (int off = 16; off; off >>= 1) s += __shfl_xor(s, off, 32);
      if (lane == 0) dst[b * CH + c * 16 + h] = 1.f / fmaxf(sqrtf(s), 1e-12f);
    }
  }
}

// ================= Kernel 2: fused attention =================
// grid = BATCH*NSLAB = 128 workgroups of 256 threads (8 wave32).
__global__ void __launch_bounds__(256)
sfa_attn(const float* __restrict__ tq, const float* __restrict__ tk,
         const float* __restrict__ tv, const float* __restrict__ invq,
         const float* __restrict__ invk, float* __restrict__ out) {
  extern __shared__ char smem[];
  bf16_t* Qs  = (bf16_t*)(smem);
  bf16_t* Ks  = (bf16_t*)(smem + KS_OFF);
  float*  Ssm = (float*)(smem);
  bf16_t* Pm  = (bf16_t*)(smem + P_OFF);
  bf16_t* Vt  = (bf16_t*)(smem);

  const int b    = blockIdx.x / NSLAB;
  const int slab = (blockIdx.x % NSLAB) * BM;
  const int tid  = threadIdx.x;
  const int wid  = tid >> 5;
  const int lane = tid & 31;
  const int half = lane >> 4;
  const int lm   = lane & 15;
  const int wm   = wid >> 2;   // 0..1 : 16-row tile
  const int wn   = wid & 3;    // 0..3 : 64-col tile (phase1) / 32-col (phase3)

  const size_t bstride = (size_t)CDIM * NWIN * HW;
  const float* qb = tq + (size_t)b * bstride;
  const float* kb = tk + (size_t)b * bstride;
  const float* vb = tv + (size_t)b * bstride;
  float*       ob = out + (size_t)b * bstride;

  const v8f vzero = {0.f, 0.f, 0.f, 0.f, 0.f, 0.f, 0.f, 0.f};

  // ---------------- Phase 1: S = Q * K^T ----------------
  v8f acc[4];
#pragma unroll
  for (int i = 0; i < 4; ++i) acc[i] = vzero;

  for (int nk = 0; nk < NWIN; ++nk) {
    // Q panel: 32 rows x 128 w, float4-vectorized, fp32 -> bf16
#pragma unroll
    for (int it = 0; it < (BM * WDIM) / 1024; ++it) {     // 4
      int idx = it * 256 + tid;
      int row = idx >> 5, w4 = (idx & 31) * 4;
      int t = slab + row;
      float4 f = *(const float4*)(qb + (size_t)((t >> 4) * NWIN + nk) * HW + (t & 15) * WDIM + w4);
      v4bf o = {(bf16_t)f.x, (bf16_t)f.y, (bf16_t)f.z, (bf16_t)f.w};
      *(v4bf*)(Qs + row * QS_STR + w4) = o;
    }
    // K panel: 256 rows x 128 w
#pragma unroll 8
    for (int it = 0; it < (CH * WDIM) / 1024; ++it) {     // 32
      int idx = it * 256 + tid;
      int r = idx >> 5, w4 = (idx & 31) * 4;
      float4 f = *(const float4*)(kb + (size_t)((r >> 4) * NWIN + nk) * HW + (r & 15) * WDIM + w4);
      v4bf o = {(bf16_t)f.x, (bf16_t)f.y, (bf16_t)f.z, (bf16_t)f.w};
      *(v4bf*)(Ks + r * KS_STR + w4) = o;
    }
    // warm caches for the next panel while this one is consumed
    if (nk + 1 < NWIN) {
#pragma unroll
      for (int it = 0; it < 4; ++it) {
        int idx = it * 256 + tid;
        int r = idx >> 2, seg = idx & 3;
        __builtin_prefetch(kb + (size_t)((r >> 4) * NWIN + nk + 1) * HW + (r & 15) * WDIM + seg * 32, 0, 3);
      }
      if (tid < 128) {
        int row = tid >> 2, seg = tid & 3;
        int t = slab + row;
        __builtin_prefetch(qb + (size_t)((t >> 4) * NWIN + nk + 1) * HW + (t & 15) * WDIM + seg * 32, 0, 3);
      }
    }
    __syncthreads();
#pragma unroll
    for (int ks = 0; ks < 4; ++ks) {
      const int k0 = ks * 32;
      v16bf af = frag_a(Qs + (wm * 16 + lm) * QS_STR + k0, half);
      v16bf bfr[4];
#pragma unroll
      for (int nt = 0; nt < 4; ++nt)
        bfr[nt] = frag_b(Ks + (wn * 64 + nt * 16 + lm) * KS_STR + k0, half);
#pragma unroll
      for (int nt = 0; nt < 4; ++nt)
        acc[nt] = __builtin_amdgcn_wmma_f32_16x16x32_bf16(
            false, af, false, bfr[nt], (short)0, acc[nt], false, false);
    }
    __syncthreads();
  }

  // spill S to LDS (C/D layout: vgpr j -> M = j + 8*half, N = lm)
#pragma unroll
  for (int nt = 0; nt < 4; ++nt)
#pragma unroll
    for (int j = 0; j < 8; ++j)
      Ssm[(wm * 16 + j + 8 * half) * CH + wn * 64 + nt * 16 + lm] = acc[nt][j];
  __syncthreads();

  // ---------------- Phase 2: scale + causal mask + softmax -> P (bf16) ----------------
  float invk8[8];
#pragma unroll
  for (int e = 0; e < 8; ++e) invk8[e] = invk[b * CH + lane * 8 + e];

#pragma unroll
  for (int i = 0; i < 4; ++i) {
    const int row  = wid * 4 + i;
    const int grow = slab + row;
    const float qs = invq[b * CH + grow];
    float x[8];
    float4 x0 = *(const float4*)(Ssm + row * CH + lane * 8);
    float4 x1 = *(const float4*)(Ssm + row * CH + lane * 8 + 4);
    x[0] = x0.x; x[1] = x0.y; x[2] = x0.z; x[3] = x0.w;
    x[4] = x1.x; x[5] = x1.y; x[6] = x1.z; x[7] = x1.w;
    float m = -1e30f;
#pragma unroll
    for (int e = 0; e < 8; ++e) {
      int col = lane * 8 + e;
      x[e] = (col <= grow) ? x[e] * qs * invk8[e] : -1e30f;
      m = fmaxf(m, x[e]);
    }
#pragma unroll
    for (int off = 16; off; off >>= 1) m = fmaxf(m, __shfl_xor(m, off, 32));
    float s = 0.f;
#pragma unroll
    for (int e = 0; e < 8; ++e) { x[e] = __expf(x[e] - m); s += x[e]; }
#pragma unroll
    for (int off = 16; off; off >>= 1) s += __shfl_xor(s, off, 32);
    const float r = 1.f / s;
#pragma unroll
    for (int e = 0; e < 8; ++e)
      Pm[row * P_STR + lane * 8 + e] = (bf16_t)(x[e] * r);
  }
  __syncthreads();   // all S reads done before Vt overwrites that region

  // ---------------- Phase 3: O = P * V, per n-panel, + residual ----------------
  for (int nk = 0; nk < NWIN; ++nk) {
    // V panel transposed into LDS: Vt[w][r]
#pragma unroll 8
    for (int it = 0; it < (CH * WDIM) / 1024; ++it) {     // 32
      int idx = it * 256 + tid;
      int r = idx >> 5, w4 = (idx & 31) * 4;
      float4 f = *(const float4*)(vb + (size_t)((r >> 4) * NWIN + nk) * HW + (r & 15) * WDIM + w4);
      Vt[(w4 + 0) * VT_STR + r] = (bf16_t)f.x;
      Vt[(w4 + 1) * VT_STR + r] = (bf16_t)f.y;
      Vt[(w4 + 2) * VT_STR + r] = (bf16_t)f.z;
      Vt[(w4 + 3) * VT_STR + r] = (bf16_t)f.w;
    }
    if (nk + 1 < NWIN) {
#pragma unroll
      for (int it = 0; it < 4; ++it) {
        int idx = it * 256 + tid;
        int r = idx >> 2, seg = idx & 3;
        __builtin_prefetch(vb + (size_t)((r >> 4) * NWIN + nk + 1) * HW + (r & 15) * WDIM + seg * 32, 0, 3);
      }
    }
    __syncthreads();

    v8f acc2[2];
    acc2[0] = vzero; acc2[1] = vzero;
#pragma unroll
    for (int ks = 0; ks < 8; ++ks) {
      const int k0 = ks * 32;
      v16bf af = frag_a(Pm + (wm * 16 + lm) * P_STR + k0, half);
      v16bf bv[2];
#pragma unroll
      for (int ct = 0; ct < 2; ++ct)
        bv[ct] = frag_b(Vt + (wn * 32 + ct * 16 + lm) * VT_STR + k0, half);
#pragma unroll
      for (int ct = 0; ct < 2; ++ct)
        acc2[ct] = __builtin_amdgcn_wmma_f32_16x16x32_bf16(
            false, af, false, bv[ct], (short)0, acc2[ct], false, false);
    }
    // epilogue: residual add (token_v re-read is L2-resident) + coalesced store
#pragma unroll
    for (int ct = 0; ct < 2; ++ct) {
#pragma unroll
      for (int j = 0; j < 8; ++j) {
        int t = slab + wm * 16 + j + 8 * half;
        int w = wn * 32 + ct * 16 + lm;
        size_t addr = (size_t)((t >> 4) * NWIN + nk) * HW + (t & 15) * WDIM + w;
        ob[addr] = acc2[ct][j] + vb[addr];
      }
    }
    __syncthreads();
  }
}

// ================= host launch =================
extern "C" void kernel_launch(void* const* d_in, const int* in_sizes, int n_in,
                              void* d_out, int out_size, void* d_ws, size_t ws_size,
                              hipStream_t stream) {
  (void)in_sizes; (void)n_in; (void)out_size; (void)ws_size;
  const float* tq = (const float*)d_in[0];
  const float* tk = (const float*)d_in[1];
  const float* tv = (const float*)d_in[2];
  float* out  = (float*)d_out;
  float* invq = (float*)d_ws;                       // BATCH*CH floats
  float* invk = (float*)d_ws + BATCH * CH;          // BATCH*CH floats

  hipFuncSetAttribute((const void*)sfa_attn,
                      hipFuncAttributeMaxDynamicSharedMemorySize, SMEM_BYTES);

  sfa_norms<<<dim3(BATCH * CDIM), dim3(256), 0, stream>>>(tq, tk, invq, invk);
  sfa_attn<<<dim3(BATCH * NSLAB), dim3(256), SMEM_BYTES, stream>>>(
      tq, tk, tv, invq, invk, out);
}
